// QRNN3DLayer_40243843563919
// MI455X (gfx1250) — matrix operand order repaired and checked by reading, not compiled, not run on hardware
//
#include <hip/hip_runtime.h>

// ---------------- CDNA5 (gfx1250, wave32) fused QRNN3D layer ----------------
// conv3d (implicit GEMM via v_wmma_f32_16x16x32_bf16) + tanh/sigmoid gates +
// fo-pool recurrence fused in one kernel. One workgroup = (batch, 2 h-rows),
// sequential loop over T with h-state in VGPRs and a 3-deep LDS ring of
// bf16 input slabs laid out [tz][hy][wx][ic] so B fragments are 2x ds_load_b128.
// Ring bases rotate in registers (no %3), tap decode is compile-time + one
// per-lane select (no /9), activations use v_exp/v_rcp fast paths.

typedef __attribute__((ext_vector_type(8)))  float        v8f;
typedef __attribute__((ext_vector_type(4)))  unsigned int v4u;
typedef __attribute__((ext_vector_type(16))) __bf16       v16bf;

union Frag { v4u u[2]; v16bf v; };

#define BATCH  4
#define CIN    16
#define TT     31
#define HH     128
#define WW     128
#define CHID   16
#define KSTEPS 14            // ceil(27 taps / 2); tap 27 zero-padded in weight LDS
#define WROW   40            // padded K-stride (ushorts) -> conflict-free A loads
#define WL_BYTES (2*KSTEPS*16*WROW*2)       // 35,840 B
#define XROW   16            // ic innermost (16 bf16 = 32 B = one B fragment row)
#define XWX    130           // w in [-1,128] inclusive
#define XHY    4             // rows r0-1 .. r0+2
#define XSLOT  (XHY*XWX*XROW)
#define XL_BYTES (3*XSLOT*2)                // 49,920 B
#define SMEM_BYTES (WL_BYTES + XL_BYTES)    // 85,760 B

__device__ __forceinline__ unsigned short f2bf(float x) {
    unsigned int u = __float_as_uint(x);
    u += 0x7fffu + ((u >> 16) & 1u);        // round-to-nearest-even
    return (unsigned short)(u >> 16);
}

__device__ __forceinline__ float fsigmoid(float x) {
    return __builtin_amdgcn_rcpf(1.0f + __expf(-x));
}
__device__ __forceinline__ float ftanh(float x) {
    // tanh(x) = 1 - 2/(exp(2x)+1); saturates correctly at +-1
    return 1.0f - 2.0f * __builtin_amdgcn_rcpf(1.0f + __expf(2.0f * x));
}

// Stage one t-slab (tz) into ring slab at dst: fp32 global -> bf16 LDS [hy][wx][ic].
// 2 rows x 128 lanes per pass (shift/mask only), wx in {128,129} handled as tail.
__device__ __forceinline__ void stage_slab(unsigned short* dst, const float* xb,
                                           int tz, int r0, int tid) {
    const bool tzok = (tz >= 0) && (tz < TT);
    const int wx  = tid & 127;
    const int rr  = tid >> 7;                // 0 or 1
    const int wr  = wx - 1;
    #pragma unroll 4
    for (int it = 0; it < 32; ++it) {
        int row = it * 2 + rr;               // hy*16 + ic, 0..63
        int ic  = row & 15;
        int hy  = row >> 4;
        int hr  = r0 - 1 + hy;
        const float* src = xb + ((size_t)ic * TT + tz) * (HH * WW) + hr * WW + wr;
        float v = 0.0f;
        if (tzok && hr >= 0 && hr < HH && wr >= 0) v = *src;
        // hint next t-slab into GL2 (one lane per wave) -> global_prefetch_b8
        if ((tid & 31) == 0 && tzok && (tz + 1) < TT && hr >= 0 && hr < HH && wr >= 0)
            __builtin_prefetch(src + HH * WW, 0, 1);
        dst[(hy * XWX + wx) * XROW + ic] = f2bf(v);
    }
    if (tid < 128) {                          // tail: wx = 128, 129
        int row = tid >> 1;
        int twx = 128 + (tid & 1);
        int ic  = row & 15;
        int hy  = row >> 4;
        int hr  = r0 - 1 + hy;
        int twr = twx - 1;                    // 127 or 128
        float v = 0.0f;
        if (tzok && hr >= 0 && hr < HH && twr < WW)
            v = xb[((size_t)ic * TT + tz) * (HH * WW) + hr * WW + twr];
        dst[(hy * XWX + twx) * XROW + ic] = f2bf(v);
    }
}

__global__ __launch_bounds__(256)
void qrnn3d_fused(const float* __restrict__ x, const float* __restrict__ Wt,
                  const float* __restrict__ bias, float* __restrict__ out) {
    extern __shared__ char smem[];
    unsigned short* wl = (unsigned short*)smem;              // weights, frag order
    unsigned short* xl = (unsigned short*)(smem + WL_BYTES); // input ring

    const int tid  = threadIdx.x;
    const int lane = tid & 31;
    const int wv   = tid >> 5;       // 8 waves
    const int n    = lane & 15;      // N position within 16-wide tile
    const int hi   = lane >> 4;      // half-wave selector
    const int w0   = wv * 16;        // wave's w-tile origin (covers all 128 w)

    const int b  = blockIdx.x >> 6;  // / (HH/2)
    const int hp = blockIdx.x & 63;
    const int r0 = hp * 2;           // this WG owns rows r0, r0+1

    const float* xb = x + (size_t)b * CIN * TT * HH * WW;

    // ---- one-time: weights -> bf16 LDS in A-fragment order [g][s][M][K] ----
    // global k = tap*16 + ic ; tap>=27 => zero pad (makes padded K inert).
    for (int e = tid; e < 2 * KSTEPS * 16 * 32; e += 256) {
        int kk = e & 31;
        int m  = (e >> 5) & 15;
        int sg = e >> 9;
        int s  = sg % KSTEPS;
        int g  = sg / KSTEPS;
        int kg  = s * 32 + kk;
        int tap = kg >> 4;
        int ic  = kg & 15;
        float v = 0.0f;
        if (tap < 27) v = Wt[(((g * 16 + m) * CIN + ic) * 27) + tap];
        wl[((g * KSTEPS + s) * 16 + m) * WROW + kk] = f2bf(v);
    }

    // per-lane bias, matching C/D layout (M = r + 8*hi)
    float bz[8], bf_[8];
    #pragma unroll
    for (int r = 0; r < 8; ++r) {
        bz[r]  = bias[r + 8 * hi];
        bf_[r] = bias[CHID + r + 8 * hi];
    }

    // initial slabs tz = -1,0,1 -> slots 2,0,1
    stage_slab(xl + 2 * XSLOT, xb, -1, r0, tid);
    stage_slab(xl + 0 * XSLOT, xb,  0, r0, tid);
    stage_slab(xl + 1 * XSLOT, xb,  1, r0, tid);

    // ring base element-offsets for slabs tz = t-1, t, t+1 (rotate per t, no %3)
    int sb0 = 2 * XSLOT, sb1 = 0, sb2 = XSLOT;

    const int wxn = (w0 + n) * XROW;                       // per-lane B column base
    const unsigned short* awb = wl + n * WROW + hi * 16;   // per-lane A row base

    float h0[8], h1[8];
    #pragma unroll
    for (int r = 0; r < 8; ++r) { h0[r] = 0.0f; h1[r] = 0.0f; }

    for (int t = 0; t < TT; ++t) {
        __syncthreads();  // staged slabs visible

        v8f accZ0 = {}, accF0 = {}, accZ1 = {}, accF1 = {};

        #pragma unroll
        for (int s = 0; s < KSTEPS; ++s) {
            // lanes 0-15 carry tap 2s, lanes 16-31 tap 2s+1; decode at compile time
            const int tap0 = 2 * s,            tap1 = 2 * s + 1;
            const int kd0  = tap0 / 9,         kd1  = tap1 / 9;        // 0..3 (3 wraps)
            const int kh0  = (tap0 % 9) / 3,   kh1  = (tap1 % 9) / 3;
            const int kw0  = tap0 % 3,         kw1  = tap1 % 3;
            const int ct0  = (kh0 * XWX + kw0) * XROW;
            const int ct1  = (kh1 * XWX + kw1) * XROW;
            const int b0s  = (kd0 == 1) ? sb1 : (kd0 == 2 ? sb2 : sb0); // kd 0/3 -> sb0
            const int b1s  = (kd1 == 1) ? sb1 : (kd1 == 2 ? sb2 : sb0);
            const int boff = (hi ? (b1s + ct1) : (b0s + ct0)) + wxn;    // one select

            Frag a0, a1, bb0, bb1;
            const unsigned short* aw0 = awb + ((0 * KSTEPS + s) * 16) * WROW;
            const unsigned short* aw1 = awb + ((1 * KSTEPS + s) * 16) * WROW;
            a0.u[0] = *(const v4u*)(aw0);        // K 16*hi..16*hi+7
            a0.u[1] = *(const v4u*)(aw0 + 8);    // K 16*hi+8..16*hi+15
            a1.u[0] = *(const v4u*)(aw1);
            a1.u[1] = *(const v4u*)(aw1 + 8);

            const unsigned short* bp = xl + boff;
            bb0.u[0] = *(const v4u*)(bp);                  // ic 0..7  (row r0)
            bb0.u[1] = *(const v4u*)(bp + 8);              // ic 8..15
            bb1.u[0] = *(const v4u*)(bp + XWX * XROW);     // row r0+1 (hy = kh+1)
            bb1.u[1] = *(const v4u*)(bp + XWX * XROW + 8);

            accZ0 = __builtin_amdgcn_wmma_f32_16x16x32_bf16(false, a0.v, false, bb0.v,
                                                            (short)0, accZ0, false, false);
            accF0 = __builtin_amdgcn_wmma_f32_16x16x32_bf16(false, a1.v, false, bb0.v,
                                                            (short)0, accF0, false, false);
            accZ1 = __builtin_amdgcn_wmma_f32_16x16x32_bf16(false, a0.v, false, bb1.v,
                                                            (short)0, accZ1, false, false);
            accF1 = __builtin_amdgcn_wmma_f32_16x16x32_bf16(false, a1.v, false, bb1.v,
                                                            (short)0, accF1, false, false);
        }

        // stage next slab first so its global-load latency overlaps the
        // transcendentals + output stores below
        if (t < TT - 1) {
            __syncthreads();                        // all reads of oldest slab done
            stage_slab(xl + sb0, xb, t + 2, r0, tid);
        }
        { int tmp = sb0; sb0 = sb1; sb1 = sb2; sb2 = tmp; }

        // gates -> fo-pool recurrence -> store h_t (C/D layout: oc=r+8*hi, w=w0+n)
        #pragma unroll
        for (int r = 0; r < 8; ++r) {
            const int oc = r + 8 * hi;
            const size_t base =
                (((size_t)b * CHID + oc) * TT + t) * (size_t)(HH * WW) + (size_t)(w0 + n);

            float z0 = ftanh(accZ0[r] + bz[r]);
            float f0 = fsigmoid(accF0[r] + bf_[r]);
            h0[r] = f0 * h0[r] + (1.0f - f0) * z0;
            out[base + (size_t)(r0 + 0) * WW] = h0[r];

            float z1 = ftanh(accZ1[r] + bz[r]);
            float f1 = fsigmoid(accF1[r] + bf_[r]);
            h1[r] = f1 * h1[r] + (1.0f - f1) * z1;
            out[base + (size_t)(r0 + 1) * WW] = h1[r];
        }
    }
}

extern "C" void kernel_launch(void* const* d_in, const int* in_sizes, int n_in,
                              void* d_out, int out_size, void* d_ws, size_t ws_size,
                              hipStream_t stream) {
    const float* x    = (const float*)d_in[0];
    const float* Wt   = (const float*)d_in[1];
    const float* bias = (const float*)d_in[2];
    float* out        = (float*)d_out;

    (void)in_sizes; (void)n_in; (void)out_size; (void)d_ws; (void)ws_size;

    hipFuncSetAttribute(reinterpret_cast<const void*>(qrnn3d_fused),
                        hipFuncAttributeMaxDynamicSharedMemorySize, SMEM_BYTES);

    qrnn3d_fused<<<dim3(BATCH * (HH / 2)), dim3(256), SMEM_BYTES, stream>>>(
        x, Wt, bias, out);
}
